// OptimizedGNNDecoder_67851893342889
// MI455X (gfx1250) — matrix operand order repaired and checked by reading, not compiled, not run on hardware
//
#include <hip/hip_runtime.h>
#include <hip/hip_bf16.h>

typedef __attribute__((ext_vector_type(16))) _Float16 v16h;
typedef __attribute__((ext_vector_type(8)))  float    v8f;

#define N_NODES  200000
#define N_GRAPHS 64
#define K_SLOTS  16
#define LATENT   128
#define HID      128
#define MAX_TILES 6316   // sum ceil(n_g/32) <= 6250 + 64

// ---------------- workspace layout (bytes) ----------------
#define WS_STARTS 0         // int[65]
#define WS_CNT    512       // int
#define WS_WL     1024      // int[MAX_TILES]
#define WS_W2T    32768     // _Float16[128*128]  (N-major: w2t[n*128+k])
#define WS_WHT    65536     // _Float16[16*128]   (N-major)
#define WS_HBIAS  69632     // float[16]
#define WS_ZW1    131072    // _Float16[64*128*16] ([g][d][k], k fastest)

union FragH { v16h v; unsigned u[8]; };

// ---- per-graph node ranges via binary search over sorted batch ----
__global__ void k_ranges(const int* __restrict__ batch, int* starts, int* cnt) {
  int t = threadIdx.x;
  if (t == 0) *cnt = 0;
  if (t > N_GRAPHS) return;
  int lo = 0, hi = N_NODES;
  while (lo < hi) { int m = (lo + hi) >> 1; if (batch[m] < t) lo = m + 1; else hi = m; }
  starts[t] = lo;
}

// ---- build worklist of 32-node tiles (graph, tile) ----
__global__ void k_worklist(const int* __restrict__ starts, int* cnt, int* wl) {
  int g = threadIdx.x;
  if (g >= N_GRAPHS) return;
  int n = starts[g + 1] - starts[g];
  int nt = (n + 31) >> 5;
  for (int t = 0; t < nt; ++t) {
    int pos = atomicAdd(cnt, 1);
    wl[pos] = (g << 16) | t;
  }
}

// ---- weight prep: transpose W2 -> f16 N-major; pack heads [Wp|Wv|0] ----
__global__ void k_prep_w(const float* __restrict__ W2, const float* __restrict__ Wp,
                         const float* __restrict__ bp, const float* __restrict__ Wv,
                         const float* __restrict__ bv,
                         _Float16* w2t, _Float16* wht, float* hbias) {
  int i = blockIdx.x * 256 + threadIdx.x;
  if (i < 128 * 128) {
    int n = i >> 7, k = i & 127;
    w2t[n * 128 + k] = (_Float16)W2[k * 128 + n];
  }
  if (i < 16 * 128) {
    int n = i >> 7, k = i & 127;
    float v = 0.f;
    if (n < 2)      v = Wp[k * 2 + n];
    else if (n < 4) v = Wv[k * 2 + (n - 2)];
    wht[n * 128 + k] = (_Float16)v;
  }
  if (i < 16) {
    float v = 0.f;
    if (i < 2)      v = bp[i];
    else if (i < 4) v = bv[i - 2];
    hbias[i] = v;
  }
}

// ---- ZW1[g][d][k] = sum_j z[g][k][j] * W1[j][d]  (f32 compute, f16 store) ----
__global__ void k_prep_zw1(const float* __restrict__ z, const float* __restrict__ W1,
                           _Float16* zw1) {
  int i = blockIdx.x * 256 + threadIdx.x;
  if (i >= N_GRAPHS * HID * K_SLOTS) return;
  int k = i & 15, d = (i >> 4) & 127, g = i >> 11;
  const float* zr = z + (g * K_SLOTS + k) * LATENT;
  float acc = 0.f;
  #pragma unroll 4
  for (int j = 0; j < LATENT; ++j) acc += zr[j] * W1[j * HID + d];
  zw1[i] = (_Float16)acc;
}

#define WMMA(A, B, C) __builtin_amdgcn_wmma_f32_16x16x32_f16(false, (A), false, (B), (short)0, (C), false, false)

__global__ __launch_bounds__(32) void k_main(
    const float* __restrict__ s,
    const float* __restrict__ b1, const float* __restrict__ b2,
    const int* __restrict__ starts, const int* __restrict__ cnt,
    const int* __restrict__ wl,
    const _Float16* __restrict__ zw1, const _Float16* __restrict__ w2t,
    const _Float16* __restrict__ wht, const float* __restrict__ hbias,
    float* __restrict__ out) {
  __shared__ _Float16 sstage[16 * 16];
  __shared__ _Float16 hstage[16 * 132];

  int bid = blockIdx.x;
  if (bid >= *cnt) return;
  int e = wl[bid];
  int g = e >> 16, t = e & 0xffff;
  int lo = starts[g], hi = starts[g + 1];
  int base = lo + t * 32;

  int lane = threadIdx.x;
  int lmod = lane & 15, lhi = lane >> 4;
  const unsigned* sstage_u = (const unsigned*)sstage;
  const unsigned* hstage_u = (const unsigned*)hstage;

  // shared read-only zero accumulator for all first-WMMAs
  v8f zacc;
  #pragma unroll
  for (int j = 0; j < 8; ++j) zacc[j] = 0.f;

  // ---------- stage s tiles, build A fragments (K=16 padded to 32) ----------
  // Rows past the valid count feed only output rows that are never stored,
  // and all of s[] is finite, so we just CLAMP the address (no divergent
  // loads, no value masking needed).
  FragH Sfrag[2];
  #pragma unroll
  for (int mt = 0; mt < 2; ++mt) {
    #pragma unroll
    for (int i = 0; i < 8; ++i) {
      int e2 = lane * 8 + i;
      int r = e2 >> 4, k = e2 & 15;
      int node = base + mt * 16 + r;
      node = node < N_NODES - 1 ? node : N_NODES - 1;
      float v = s[node * 16 + k];
      sstage[r * 16 + k] = (_Float16)v;
    }
    __syncthreads();
    FragH f;
    #pragma unroll
    for (int vv = 0; vv < 4; ++vv) {
      int K = lhi * 8 + vv * 2;
      f.u[vv] = sstage_u[(lmod * 16 + K) >> 1];
    }
    #pragma unroll
    for (int vv = 4; vv < 8; ++vv) f.u[vv] = 0;
    Sfrag[mt] = f;
    __syncthreads();
  }

  // ---------- layer 1 (fused slot-sum): C = S x ZW1_g  (bias in epilogue) ----
  // A-fragment K=16..31 is hard zero, so B elements 8-15 (K=16..31) may hold
  // ANY finite data: 0*finite == 0. All lanes load the same zw1 dwords
  // (the 8 dwords of row lane&15, rotated by 4 for lanes 16-31) -> uniform, branch-free loads.
  v8f C[2][8];
  const unsigned* zw1u = (const unsigned*)(zw1 + (size_t)g * HID * K_SLOTS);
  #pragma unroll
  for (int nt = 0; nt < 8; ++nt) {
    int d = nt * 16 + lmod;
    FragH B;
    #pragma unroll
    for (int vv = 0; vv < 8; ++vv) B.u[vv] = zw1u[d * 8 + ((vv + lhi * 4) & 7)];
    C[0][nt] = WMMA(Sfrag[0].v, B.v, zacc);
    C[1][nt] = WMMA(Sfrag[1].v, B.v, zacc);
  }

  // ---------- bias+relu, restage, load A fragments for layer 2 ----------
  FragH A2[2][4];
  float bias1[8];
  #pragma unroll
  for (int nt = 0; nt < 8; ++nt) bias1[nt] = b1[nt * 16 + lmod];
  #pragma unroll
  for (int mt = 0; mt < 2; ++mt) {
    #pragma unroll
    for (int nt = 0; nt < 8; ++nt)
      #pragma unroll
      for (int j = 0; j < 8; ++j) {
        float v = __builtin_fmaxf(C[mt][nt][j] + bias1[nt], 0.f);
        hstage[(j + 8 * lhi) * 132 + nt * 16 + lmod] = (_Float16)v;
      }
    __syncthreads();
    #pragma unroll
    for (int kc = 0; kc < 4; ++kc) {
      FragH f;
      #pragma unroll
      for (int vv = 0; vv < 8; ++vv) {
        int K = kc * 32 + lhi * 8 + (vv & 3) * 2 + (vv >> 2) * 16;
        f.u[vv] = hstage_u[(lmod * 132 + K) >> 1];
      }
      A2[mt][kc] = f;
    }
    __syncthreads();
  }

  // ---------- layer 2: C = A2 x W2  (bias in epilogue) ----------
  const unsigned* w2u = (const unsigned*)w2t;
  #pragma unroll
  for (int nt = 0; nt < 8; ++nt) {
    v8f c0 = zacc, c1 = zacc;
    int n = nt * 16 + lmod;
    #pragma unroll
    for (int kc = 0; kc < 4; ++kc) {
      int Kb = kc * 32 + lhi * 8;
      FragH B;
      #pragma unroll
      for (int vv = 0; vv < 8; ++vv) B.u[vv] = w2u[n * 64 + (Kb >> 1) + (vv & 3) + (vv >> 2) * 8];
      c0 = WMMA(A2[0][kc].v, B.v, c0);
      c1 = WMMA(A2[1][kc].v, B.v, c1);
    }
    C[0][nt] = c0; C[1][nt] = c1;
  }

  // ---------- bias+relu, restage, A fragments for heads ----------
  float bias2[8];
  #pragma unroll
  for (int nt = 0; nt < 8; ++nt) bias2[nt] = b2[nt * 16 + lmod];
  #pragma unroll
  for (int mt = 0; mt < 2; ++mt) {
    #pragma unroll
    for (int nt = 0; nt < 8; ++nt)
      #pragma unroll
      for (int j = 0; j < 8; ++j) {
        float v = __builtin_fmaxf(C[mt][nt][j] + bias2[nt], 0.f);
        hstage[(j + 8 * lhi) * 132 + nt * 16 + lmod] = (_Float16)v;
      }
    __syncthreads();
    #pragma unroll
    for (int kc = 0; kc < 4; ++kc) {
      FragH f;
      #pragma unroll
      for (int vv = 0; vv < 8; ++vv) {
        int K = kc * 32 + lhi * 8 + (vv & 3) * 2 + (vv >> 2) * 16;
        f.u[vv] = hstage_u[(lmod * 132 + K) >> 1];
      }
      A2[mt][kc] = f;
    }
    __syncthreads();
  }

  // ---------- heads: one 16-wide B = [Wp | Wv | 0]  (bias at store) ----------
  const unsigned* whu = (const unsigned*)wht;
  float hb = hbias[lmod];
  v8f a0 = zacc, a1 = zacc;
  #pragma unroll
  for (int kc = 0; kc < 4; ++kc) {
    int Kb = kc * 32 + lhi * 8;
    FragH B;
    #pragma unroll
    for (int vv = 0; vv < 8; ++vv) B.u[vv] = whu[lmod * 64 + (Kb >> 1) + (vv & 3) + (vv >> 2) * 8];
    a0 = WMMA(A2[0][kc].v, B.v, a0);
    a1 = WMMA(A2[1][kc].v, B.v, a1);
  }

  // ---------- store: cols 0,1 tanh (pos), cols 2,3 linear (vel) ----------
  #pragma unroll
  for (int mt = 0; mt < 2; ++mt) {
    v8f a = mt ? a1 : a0;
    int vcnt = hi - (base + mt * 16);
    #pragma unroll
    for (int j = 0; j < 8; ++j) {
      int M = j + 8 * lhi;
      if (lmod < 4 && M < vcnt) {
        float v = a[j] + hb;
        if (lmod < 2) v = tanhf(v);
        out[(base + mt * 16 + M) * 4 + lmod] = v;
      }
    }
  }
}

extern "C" void kernel_launch(void* const* d_in, const int* in_sizes, int n_in,
                              void* d_out, int out_size, void* d_ws, size_t ws_size,
                              hipStream_t stream) {
  (void)in_sizes; (void)n_in; (void)out_size; (void)ws_size;
  const float* z     = (const float*)d_in[0];
  const float* s     = (const float*)d_in[1];
  const int*   batch = (const int*)d_in[2];
  const float* W1    = (const float*)d_in[3];
  const float* b1    = (const float*)d_in[4];
  const float* W2    = (const float*)d_in[5];
  const float* b2    = (const float*)d_in[6];
  const float* Wp    = (const float*)d_in[7];
  const float* bp    = (const float*)d_in[8];
  const float* Wv    = (const float*)d_in[9];
  const float* bv    = (const float*)d_in[10];
  float* out = (float*)d_out;

  char* ws = (char*)d_ws;
  int*       starts = (int*)(ws + WS_STARTS);
  int*       cnt    = (int*)(ws + WS_CNT);
  int*       wl     = (int*)(ws + WS_WL);
  _Float16*  w2t    = (_Float16*)(ws + WS_W2T);
  _Float16*  wht    = (_Float16*)(ws + WS_WHT);
  float*     hbias  = (float*)(ws + WS_HBIAS);
  _Float16*  zw1    = (_Float16*)(ws + WS_ZW1);

  k_ranges<<<1, 128, 0, stream>>>(batch, starts, cnt);
  k_prep_w<<<64, 256, 0, stream>>>(W2, Wp, bp, Wv, bv, w2t, wht, hbias);
  k_prep_zw1<<<(N_GRAPHS * HID * K_SLOTS + 255) / 256, 256, 0, stream>>>(z, W1, zw1);
  k_worklist<<<1, 64, 0, stream>>>(starts, cnt, wl);
  k_main<<<MAX_TILES, 32, 0, stream>>>(s, b1, b2, starts, cnt, wl,
                                       zw1, w2t, wht, hbias, out);
}
